// Model_68281390071881
// MI455X (gfx1250) — matrix operand order repaired and checked by reading, not compiled, not run on hardware
//
#include <hip/hip_runtime.h>

// ---------------- model constants ----------------
#define BB   2
#define SS   4096
#define HH   12
#define DH   64
#define HID  768
#define FFN_ 3072
#define NL   4
#define BS_  64
#define NB_  64
#define LL   8
#define MS   (BB*SS)          // 8192 rows

typedef __attribute__((ext_vector_type(16))) __bf16 v16bf;
typedef __attribute__((ext_vector_type(8)))  float  v8f;

union AFrag { v16bf v; uint4 q[2]; unsigned int u[8]; };
union CFrag { v8f  v; float f[8]; };

__device__ __forceinline__ unsigned short f2bf(float x) {
    __bf16 h = (__bf16)x;
    return __builtin_bit_cast(unsigned short, h);
}

__device__ __forceinline__ float gelu_tanh(float x) {
    float x3 = x * x * x;
    return 0.5f * x * (1.0f + tanhf(0.7978845608028654f * (x + 0.044715f * x3)));
}

// A fragment (16x32 bf16, M=lane&15): contiguous K chunk of 16 at koff=(lane>>4)*16
__device__ __forceinline__ v16bf load_frag_A(const unsigned short* base, int ld, int kb, int lane) {
    int m = lane & 15, koff = (lane >> 4) * 16;
    AFrag a;
    const unsigned short* p = base + (size_t)m * ld + kb + koff;
    a.q[0] = *reinterpret_cast<const uint4*>(p);
    a.q[1] = *reinterpret_cast<const uint4*>(p + 8);
    return a.v;
}

// B fragment (32x16 bf16, N=lane&15): contiguous K chunk of 16 at koff=(lane>>4)*16.
// Source is stored N-major: src[n*ld + k]  (i.e. B transposed / K^T natural layout)
__device__ __forceinline__ v16bf load_frag_B(const unsigned short* base, int ld, int kb, int lane) {
    int n = lane & 15, koff = (lane >> 4) * 16;
    AFrag a;
    const unsigned short* p = base + (size_t)n * ld + kb + koff;
    a.q[0] = *reinterpret_cast<const uint4*>(p);
    a.q[1] = *reinterpret_cast<const uint4*>(p + 8);
    return a.v;
}

// B fragment where source is K-major (V: src[k*ld + n]) -> pack pairs on the fly
__device__ __forceinline__ v16bf load_frag_Bt(const unsigned short* base, int ld, int kb, int lane) {
    int n = lane & 15, koff = (lane >> 4) * 16;
    AFrag a;
    const unsigned short* p = base + n + (size_t)(kb + koff) * ld;
#pragma unroll
    for (int j = 0; j < 8; ++j) {
        unsigned int lo = p[(size_t)(2 * j) * ld];
        unsigned int hi = p[(size_t)(2 * j + 1) * ld];
        a.u[j] = lo | (hi << 16);
    }
    return a.v;
}

#define WMMA_BF16(a, b, c) \
    __builtin_amdgcn_wmma_f32_16x16x32_bf16(false, (a), false, (b), (short)0, (c), false, false)

// ---------------- weight convert: w[K][N] f32 -> wT[N][K] bf16 ----------------
__global__ void conv_wT_kernel(const float* __restrict__ w, unsigned short* __restrict__ wT,
                               int K, int N) {
    size_t i = (size_t)blockIdx.x * blockDim.x + threadIdx.x;
    size_t total = (size_t)K * N;
    if (i >= total) return;
    int n = (int)(i / K);
    int k = (int)(i % K);
    wT[i] = f2bf(w[(size_t)k * N + n]);
}

// ---------------- embedding gather ----------------
__global__ void embed_kernel(const int* __restrict__ ids, const float* __restrict__ emb,
                             const float* __restrict__ pos, float* __restrict__ out) {
    size_t i = (size_t)blockIdx.x * blockDim.x + threadIdx.x;
    if (i >= (size_t)MS * HID) return;
    int row = (int)(i / HID), c = (int)(i % HID);
    int tok = ids[row];
    out[i] = emb[(size_t)tok * HID + c] + pos[(size_t)(row % SS) * HID + c];
}

// ---------------- residual + layernorm (row per block) ----------------
__global__ void add_ln_kernel(const float* __restrict__ x, const float* __restrict__ y,
                              const float* __restrict__ g, const float* __restrict__ bta,
                              float* __restrict__ outF, unsigned short* __restrict__ outB) {
    const int N = HID;
    int row = blockIdx.x, t = threadIdx.x;
    __shared__ float buf[HID];
    __shared__ float red[256], red2[256];
    float s = 0.f, s2 = 0.f;
    for (int i = t; i < N; i += 256) {
        float v = x[(size_t)row * N + i];
        if (y) v += y[(size_t)row * N + i];
        buf[i] = v; s += v; s2 += v * v;
    }
    red[t] = s; red2[t] = s2;
    __syncthreads();
    for (int o = 128; o > 0; o >>= 1) {
        if (t < o) { red[t] += red[t + o]; red2[t] += red2[t + o]; }
        __syncthreads();
    }
    float mean = red[0] / N;
    float var  = red2[0] / N - mean * mean;
    float r = rsqrtf(var + 1e-12f);
    for (int i = t; i < N; i += 256) {
        float v = (buf[i] - mean) * r * g[i] + bta[i];
        outF[(size_t)row * N + i] = v;
        outB[(size_t)row * N + i] = f2bf(v);
    }
}

// ---------------- register-blocked WMMA GEMM ----------------
// 128 threads = 4 waves, one 64x64 macro-tile per wave (4x4 WMMA tiles).
// __launch_bounds__(128,1): 1 wave/SIMD occupancy target so the allocator can
// hold 16 f32 accumulators (128 VGPRs) + 4 B fragments (32) + 1 A fragment (8)
// in registers with no scratch spills. ~1 vector load per WMMA.
// C[M][N] = A[M][K] (bf16) * WT[N][K]^T (bf16) + bias, f32 accumulate.
__global__ __launch_bounds__(128, 1)
void gemm_wmma_kernel(const unsigned short* __restrict__ A,
                      const unsigned short* __restrict__ WT,
                      const float* __restrict__ bias,
                      float* __restrict__ outF, unsigned short* __restrict__ outB,
                      int M, int N, int K, int act) {
    int lane = threadIdx.x & 31, wave = threadIdx.x >> 5;
    int tilesN = N >> 6;                 // 64-wide macro-tiles
    int ntiles = (M >> 6) * tilesN;
    int tile = blockIdx.x * 4 + wave;
    if (tile >= ntiles) return;
    int tm = tile / tilesN, tn = tile % tilesN;
    const unsigned short* Ab = A  + (size_t)(tm * 64) * K;
    const unsigned short* Bb = WT + (size_t)(tn * 64) * K;

    CFrag c[4][4];
#pragma unroll
    for (int i = 0; i < 4; ++i)
#pragma unroll
        for (int j = 0; j < 4; ++j)
#pragma unroll
            for (int r = 0; r < 8; ++r) c[i][j].f[r] = 0.f;

    for (int kb = 0; kb < K; kb += 32) {
        v16bf b[4];
#pragma unroll
        for (int j = 0; j < 4; ++j)
            b[j] = load_frag_B(Bb + (size_t)(j * 16) * K, K, kb, lane);
#pragma unroll
        for (int i = 0; i < 4; ++i) {
            v16bf a = load_frag_A(Ab + (size_t)(i * 16) * K, K, kb, lane);
#pragma unroll
            for (int j = 0; j < 4; ++j)
                c[i][j].v = WMMA_BF16(a, b[j], c[i][j].v);
        }
    }

    int nl = lane & 15, mb = (lane >> 4) * 8;
#pragma unroll
    for (int i = 0; i < 4; ++i) {
#pragma unroll
        for (int j = 0; j < 4; ++j) {
            int col = tn * 64 + j * 16 + nl;
            float bsv = bias ? bias[col] : 0.f;
#pragma unroll
            for (int r = 0; r < 8; ++r) {
                int row = tm * 64 + i * 16 + mb + r;
                float v = c[i][j].f[r] + bsv;
                if (act == 1) v = gelu_tanh(v);
                if (outF) outF[(size_t)row * N + col] = v;
                if (outB) outB[(size_t)row * N + col] = f2bf(v);
            }
        }
    }
}

// ---------------- block-sparse flash attention ----------------
// grid = B*H*NB blocks of 128 threads (4 waves x 16 query rows).
// Middle blocks gather LL key-blocks via band_idx; edge blocks (0, NB-1) stream all NB blocks.
// Output (bf16) overwrites the Q buffer in place.
__global__ __launch_bounds__(128, 1)
void attn_kernel(const unsigned short* __restrict__ Q,
                 const unsigned short* __restrict__ Kb,
                 const unsigned short* __restrict__ Vb,
                 const float* __restrict__ mask,
                 const int* __restrict__ band,
                 unsigned short* __restrict__ Ob) {
    __shared__ __align__(16) float          sc_s[4][16][64];
    __shared__ __align__(16) unsigned short pr_s[4][16][64];
    __shared__ float mi_s[4][16], li_s[4][16], scl_s[4][16];

    int lane = threadIdx.x & 31, w = threadIdx.x >> 5;
    int nb = blockIdx.x % NB_;
    int h  = (blockIdx.x / NB_) % HH;
    int b  = blockIdx.x / (NB_ * HH);
    int nl = lane & 15, mb = (lane >> 4) * 8;
    bool edge = (nb == 0) || (nb == NB_ - 1);
    int nkb = edge ? NB_ : LL;

    if (lane < 16) { mi_s[w][lane] = -3.0e38f; li_s[w][lane] = 0.f; }

    CFrag o[4];
#pragma unroll
    for (int t = 0; t < 4; ++t)
#pragma unroll
        for (int r = 0; r < 8; ++r) o[t].f[r] = 0.f;

    const unsigned short* Qbase =
        Q + (size_t)(b * SS + nb * BS_ + w * 16) * HID + h * DH;

    for (int j = 0; j < nkb; ++j) {
        int kblk = edge ? j : band[(nb - 1) * LL + j];
        const unsigned short* Kbase = Kb + (size_t)(b * SS + kblk * BS_) * HID + h * DH;

        // ---- scores: S = Q (16x64) * K^T (64 keys) ----
        for (int tn = 0; tn < 4; ++tn) {
            CFrag s;
#pragma unroll
            for (int r = 0; r < 8; ++r) s.f[r] = 0.f;
            for (int kb2 = 0; kb2 < DH; kb2 += 32) {
                v16bf a  = load_frag_A(Qbase, HID, kb2, lane);
                v16bf bb = load_frag_B(Kbase + (size_t)(tn * 16) * HID, HID, kb2, lane);
                s.v = WMMA_BF16(a, bb, s.v);
            }
#pragma unroll
            for (int r = 0; r < 8; ++r)
                sc_s[w][mb + r][tn * 16 + nl] = s.f[r] * 0.125f;
        }
        __syncthreads();

        // ---- online softmax (row per lane 0..15) ----
        if (lane < 16) {
            int r = lane;
            float mold = mi_s[w][r], rmax = mold;
            for (int cdx = 0; cdx < 64; ++cdx) {
                float msk = mask[(size_t)b * SS + kblk * BS_ + cdx];
                float v = sc_s[w][r][cdx] + (msk - 1.0f) * 1.0e9f;
                sc_s[w][r][cdx] = v;
                rmax = fmaxf(rmax, v);
            }
            float so = expf(mold - rmax);
            float lsum = 0.f;
            for (int cdx = 0; cdx < 64; ++cdx) {
                float p = expf(sc_s[w][r][cdx] - rmax);
                pr_s[w][r][cdx] = f2bf(p);
                lsum += p;
            }
            li_s[w][r] = so * li_s[w][r] + lsum;
            mi_s[w][r] = rmax;
            scl_s[w][r] = so;
        }
        __syncthreads();

        // ---- rescale accumulators, then O += P (16x64) * V (64 x 64dh) ----
#pragma unroll
        for (int t = 0; t < 4; ++t)
#pragma unroll
            for (int r = 0; r < 8; ++r) o[t].f[r] *= scl_s[w][mb + r];

        const unsigned short* Vbase = Vb + (size_t)(b * SS + kblk * BS_) * HID + h * DH;
        const unsigned short* Pbase = &pr_s[w][0][0];
        for (int on = 0; on < 4; ++on) {
            for (int kb2 = 0; kb2 < 64; kb2 += 32) {
                v16bf a  = load_frag_A(Pbase, 64, kb2, lane);
                v16bf bb = load_frag_Bt(Vbase + on * 16, HID, kb2, lane);
                o[on].v = WMMA_BF16(a, bb, o[on].v);
            }
        }
        __syncthreads();
    }

    // ---- final normalize and write (overwrites Q region we own) ----
#pragma unroll
    for (int on = 0; on < 4; ++on)
#pragma unroll
        for (int r = 0; r < 8; ++r) {
            float denom = li_s[w][mb + r];
            float v = o[on].f[r] / denom;
            Ob[(size_t)(b * SS + nb * BS_ + w * 16 + mb + r) * HID + h * DH + on * 16 + nl] =
                f2bf(v);
        }
}

// ---------------- classifier head (tiny, scalar f32) ----------------
__global__ void head_kernel(const float* __restrict__ x,
                            const float* __restrict__ w1, const float* __restrict__ b1,
                            const float* __restrict__ w2, const float* __restrict__ b2,
                            const float* __restrict__ w3, const float* __restrict__ b3,
                            float* __restrict__ out) {
    int b = blockIdx.x, t = threadIdx.x;
    __shared__ float cls[HID];
    __shared__ float h1[512];
    __shared__ float h2[256];
    __shared__ float lg[6];
    __shared__ float lse;
    for (int i = t; i < HID; i += 256) cls[i] = x[(size_t)(b * SS) * HID + i];
    __syncthreads();
    for (int j = t; j < 512; j += 256) {
        float s = b1[j];
        for (int i = 0; i < HID; ++i) s += cls[i] * w1[(size_t)i * 512 + j];
        h1[j] = fmaxf(s, 0.f);
    }
    __syncthreads();
    if (t < 256) {
        float s = b2[t];
        for (int i = 0; i < 512; ++i) s += h1[i] * w2[(size_t)i * 256 + t];
        h2[t] = fmaxf(s, 0.f);
    }
    __syncthreads();
    if (t < 6) {
        float s = b3[t];
        for (int i = 0; i < 256; ++i) s += h2[i] * w3[(size_t)i * 6 + t];
        lg[t] = s;
    }
    __syncthreads();
    if (t == 0) {
        float m = lg[0];
        for (int j = 1; j < 6; ++j) m = fmaxf(m, lg[j]);
        float s = 0.f;
        for (int j = 0; j < 6; ++j) s += expf(lg[j] - m);
        lse = m + logf(s);
    }
    __syncthreads();
    if (t < 6) out[b * 6 + t] = lg[t] - lse;
}

// ---------------- orchestration ----------------
static inline int cdiv(size_t a, size_t b) { return (int)((a + b - 1) / b); }

extern "C" void kernel_launch(void* const* d_in, const int* in_sizes, int n_in,
                              void* d_out, int out_size, void* d_ws, size_t ws_size,
                              hipStream_t stream) {
    (void)in_sizes; (void)n_in; (void)out_size; (void)ws_size;

    const int*   sent_id = (const int*)d_in[0];
    const float* mask    = (const float*)d_in[1];
    const float* emb     = (const float*)d_in[2];
    const float* pos     = (const float*)d_in[3];
    const float* eln_g   = (const float*)d_in[4];
    const float* eln_b   = (const float*)d_in[5];
    // layers: 6 + 16*l + {wq,bq,wk,bk,wv,bv,wo,bo,ln1_g,ln1_b,w1,b1,w2,b2,ln2_g,ln2_b}
    const float* h_w1 = (const float*)d_in[70];
    const float* h_b1 = (const float*)d_in[71];
    const float* h_w2 = (const float*)d_in[72];
    const float* h_b2 = (const float*)d_in[73];
    const float* h_w3 = (const float*)d_in[74];
    const float* h_b3 = (const float*)d_in[75];
    const int*   band = (const int*)d_in[76];

    // workspace carve-up
    char* p = (char*)d_ws;
    auto alloc = [&](size_t bytes) { char* r = p; p += (bytes + 255) & ~(size_t)255; return r; };
    float*          xf = (float*)alloc((size_t)MS * HID * 4);
    unsigned short* xb = (unsigned short*)alloc((size_t)MS * HID * 2);
    unsigned short* qb = (unsigned short*)alloc((size_t)MS * HID * 2);
    unsigned short* kb = (unsigned short*)alloc((size_t)MS * HID * 2);
    unsigned short* vb = (unsigned short*)alloc((size_t)MS * HID * 2);
    float*          yf = (float*)alloc((size_t)MS * HID * 4);
    unsigned short* hb = (unsigned short*)alloc((size_t)MS * FFN_ * 2);
    unsigned short* wt = (unsigned short*)alloc((size_t)FFN_ * HID * 2);

    auto conv = [&](const float* w, int K, int N) {
        conv_wT_kernel<<<cdiv((size_t)K * N, 256), 256, 0, stream>>>(w, wt, K, N);
    };
    auto gemm = [&](const unsigned short* A, const float* bias, float* oF,
                    unsigned short* oB, int M, int N, int K, int act) {
        int tiles = (M / 64) * (N / 64);
        gemm_wmma_kernel<<<cdiv(tiles, 4), 128, 0, stream>>>(A, wt, bias, oF, oB, M, N, K, act);
    };

    // embedding + LN
    embed_kernel<<<cdiv((size_t)MS * HID, 256), 256, 0, stream>>>(sent_id, emb, pos, yf);
    add_ln_kernel<<<MS, 256, 0, stream>>>(yf, nullptr, eln_g, eln_b, xf, xb);

    for (int l = 0; l < NL; ++l) {
        const float** L = (const float**)&d_in[6 + 16 * l];
        const float *wq = L[0], *bq = L[1], *wk = L[2], *bkk = L[3], *wv = L[4], *bv = L[5];
        const float *wo = L[6], *bo = L[7], *ln1g = L[8], *ln1b = L[9];
        const float *w1 = L[10], *b1 = L[11], *w2 = L[12], *b2 = L[13];
        const float *ln2g = L[14], *ln2b = L[15];

        conv(wq, HID, HID); gemm(xb, bq,  nullptr, qb, MS, HID, HID, 0);
        conv(wk, HID, HID); gemm(xb, bkk, nullptr, kb, MS, HID, HID, 0);
        conv(wv, HID, HID); gemm(xb, bv,  nullptr, vb, MS, HID, HID, 0);

        attn_kernel<<<BB * HH * NB_, 128, 0, stream>>>(qb, kb, vb, mask, band, qb);

        conv(wo, HID, HID); gemm(qb, bo, yf, nullptr, MS, HID, HID, 0);
        add_ln_kernel<<<MS, 256, 0, stream>>>(xf, yf, ln1g, ln1b, xf, xb);

        conv(w1, HID, FFN_); gemm(xb, b1, nullptr, hb, MS, FFN_, HID, 1);
        conv(w2, FFN_, HID); gemm(hb, b2, yf, nullptr, MS, HID, FFN_, 0);
        add_ln_kernel<<<MS, 256, 0, stream>>>(xf, yf, ln2g, ln2b, xf, xb);
    }

    head_kernel<<<BB, 256, 0, stream>>>(xf, h_w1, h_b1, h_w2, h_b2, h_w3, h_b3, (float*)d_out);
}
